// AUSTGN_14998025798424
// MI455X (gfx1250) — matrix operand order repaired and checked by reading, not compiled, not run on hardware
//
#include <hip/hip_runtime.h>
#include <hip/hip_bf16.h>
#include <math.h>

// Problem constants
#define B_   256
#define T_   64
#define S_   10
#define H_   128
#define IPT_ 128
#define Q_   128

#define NB    4              // batches per workgroup
#define NROW  40             // NB * S_
#define NTHR  512            // scan kernel threads (16 wave32)
#define ASTRIDE 134          // padded LDS row stride (halves) -> bank-conflict-free A-frag reads

typedef __attribute__((ext_vector_type(16))) __bf16 v16bf;
typedef __attribute__((ext_vector_type(8)))  float  v8f;

union ABfrag { unsigned int u[8]; v16bf v; };

// ---- workspace layout (bytes) ----
#define WPACK_UINTS (17u * 8192u)                       // 17 weight mats, frag-packed
#define WPACK_BYTES ((size_t)WPACK_UINTS * 4u)          // 557056
#define QP_BYTES    ((size_t)B_ * T_ * S_ * H_ * 2u)    // bf16
#define QP_OFF      (WPACK_BYTES)
#define PK_OFF      (QP_OFF + QP_BYTES)
#define PV_OFF      (PK_OFF + QP_BYTES)

// ---- output layout (floats) ----
#define OUT_H    ((size_t)B_ * T_ * S_ * H_)
#define OUT_C    (OUT_H + (size_t)B_ * S_ * H_)
#define OUT_LOSS (OUT_C + (size_t)B_ * S_ * H_)

// ---- scan kernel LDS layout (bytes) ----
#define OFF_HBUF  0u        // float[40*128]
#define OFF_CBUF  20480u    // float[40*128]
#define OFF_PH    40960u    // float[4*40*128]  (reused for pk/pv GEMM outputs)
#define OFF_PX    122880u   // float[8*4*128]
#define OFF_CTX   139264u   // float[40*128]
#define OFF_AT    159744u   // float[40*128]
#define OFF_SC    180224u   // float[40*64]
#define OFF_ABUF  190464u   // ushort[48*134]
#define OFF_XTB   203328u   // ushort[16*134]
#define OFF_NV    207616u   // int[64]
#define OFF_LSUM  207872u   // float[4]
#define OFF_STT   207888u   // float[4]
#define OFF_SDT   207904u   // float[4]
#define OFF_SY    207920u   // float[4]
#define OFF_SKL   207936u   // int[4]
#define SMEM_SCAN 208064u

// ---------------- helpers ----------------
__device__ __forceinline__ float bf2f(unsigned short v) {
  return __uint_as_float(((unsigned int)v) << 16);
}
__device__ __forceinline__ float bflo(unsigned int u) { return __uint_as_float(u << 16); }
__device__ __forceinline__ float bfhi(unsigned int u) { return __uint_as_float(u & 0xffff0000u); }
__device__ __forceinline__ unsigned short f2bf(float f) {   // round-to-nearest-even
  unsigned int u = __float_as_uint(f);
  u += 0x7fffu + ((u >> 16) & 1u);
  return (unsigned short)(u >> 16);
}
__device__ __forceinline__ float sigmf(float x) { return 1.0f / (1.0f + __expf(-x)); }

// Wave-level 16x16 (M,N) x K=128 bf16 WMMA tile.
//  A: LDS bf16, row-major with `astride` halves per row (rows mbase..mbase+15 must be valid).
//  wmat: fragment-packed weight matrix [kt][nt][v][lane] dwords (K=128, N=128).
__device__ __forceinline__ v8f wmma_gemm_128(const unsigned short* __restrict__ A, int astride,
                                             int mbase, const unsigned int* __restrict__ wmat,
                                             int nt, int lane) {
  v8f acc = {};
  const int m = mbase + (lane & 15);
  const int hsel = (lane >> 4) & 1;
  const unsigned int* arow = (const unsigned int*)(A + (size_t)m * astride);
#pragma unroll
  for (int kt = 0; kt < 4; ++kt) {
    ABfrag fa, fb;
#pragma unroll
    for (int v = 0; v < 8; ++v) {
      // 16-bit A layout: lanes0-15: V0-3 -> K0..7, V4-7 -> K16..23 ; lanes16-31: +8
      int kk = ((v < 4) ? (2 * v) : (16 + 2 * (v - 4))) + (hsel ? 8 : 0);
      fa.u[v] = arow[(kt * 32 + kk) >> 1];
    }
    const unsigned int* bblk = wmat + (size_t)(kt * 8 + nt) * 256;
#pragma unroll
    for (int v = 0; v < 8; ++v) fb.u[v] = bblk[v * 32 + lane];   // coalesced 128B/vgpr
    acc = __builtin_amdgcn_wmma_f32_16x16x32_bf16(false, fa.v, false, fb.v,
                                                  (short)0, acc, false, false);
  }
  return acc;
}

// Scatter a 16x16 f32 D tile into an LDS fp32 buffer (row guard for padded M).
__device__ __forceinline__ void store_tile(float* __restrict__ O, int ostride, int mbase,
                                           int rows, int nbase, v8f acc, int lane) {
  const int n = nbase + (lane & 15);
  const int mo = (lane >> 4) ? 8 : 0;
#pragma unroll
  for (int r = 0; r < 8; ++r) {
    int m = mbase + mo + r;
    if (m < rows) O[(size_t)m * ostride + n] = acc[r];
  }
}

// ---------------- kernel 1: weight fragment packing (+loss zero) ----------------
// mats: 0=Wa, 1..4=Wattn[0..3], 5..8=Wh[0..3], 9..16=Wx[0..7]
__global__ void pack_kernel(const float* __restrict__ Wa, const float* __restrict__ Wattn,
                            const float* __restrict__ Wh, const float* __restrict__ Wx,
                            unsigned int* __restrict__ wpack, float* __restrict__ out) {
  const unsigned int gid = blockIdx.x * 256u + threadIdx.x;
  if (gid == 0) out[OUT_LOSS] = 0.0f;
  if (gid >= WPACK_UINTS) return;
  const int mat = gid >> 13;
  const int rem = gid & 8191;
  const int kt = rem >> 11;
  const int nt = (rem >> 8) & 7;
  const int v  = (rem >> 5) & 7;
  const int lane = rem & 31;
  const int n = nt * 16 + (lane & 15);
  const int half = (lane >> 4) & 1;
  const int k = kt * 32 + ((v < 4) ? (2 * v) : (16 + 2 * (v - 4))) + (half ? 8 : 0);
  const float* W;
  if (mat == 0)      W = Wa;
  else if (mat < 5)  W = Wattn + (size_t)(mat - 1) * (H_ * H_);
  else if (mat < 9)  W = Wh    + (size_t)(mat - 5) * (H_ * H_);
  else               W = Wx    + (size_t)(mat - 9) * (IPT_ * H_);
  unsigned int lo = f2bf(W[(size_t)k * H_ + n]);
  unsigned int hi = f2bf(W[(size_t)(k + 1) * H_ + n]);
  wpack[gid] = lo | (hi << 16);
}

// ---------------- kernel 2: qp = (x_q @ Wa + ba) @ Wattn0 + battn0 -> bf16 ----------------
__global__ __launch_bounds__(256) void qp_kernel(const float* __restrict__ xq,
                                                 const float* __restrict__ ba,
                                                 const float* __restrict__ battn,
                                                 const unsigned int* __restrict__ wpack,
                                                 unsigned short* __restrict__ qp) {
  __shared__ unsigned short xa[16 * ASTRIDE];
  __shared__ unsigned short tb[16 * ASTRIDE];
  const int tid = threadIdx.x, lane = tid & 31, wave = tid >> 5;  // 8 waves = 8 N tiles
  const size_t tile = blockIdx.x;                                  // 10240 row-tiles of 16
  const float* xrow = xq + tile * 16 * Q_;
  for (int i = tid; i < 16 * Q_; i += 256) {
    int m = i >> 7, k = i & 127;
    xa[m * ASTRIDE + k] = f2bf(xrow[i]);
  }
  __syncthreads();
  {
    v8f acc = wmma_gemm_128(xa, ASTRIDE, 0, wpack /*Wa*/, wave, lane);
    int n = wave * 16 + (lane & 15);
    int mo = (lane >> 4) ? 8 : 0;
    float bias = ba[n];
#pragma unroll
    for (int r = 0; r < 8; ++r) tb[(r + mo) * ASTRIDE + n] = f2bf(acc[r] + bias);
  }
  __syncthreads();
  {
    v8f acc = wmma_gemm_128(tb, ASTRIDE, 0, wpack + 8192 /*Wattn0*/, wave, lane);
    int n = wave * 16 + (lane & 15);
    int mo = (lane >> 4) ? 8 : 0;
    float bias = battn[n];
    unsigned short* qrow = qp + tile * 16 * H_;
#pragma unroll
    for (int r = 0; r < 8; ++r) qrow[(size_t)(r + mo) * H_ + n] = f2bf(acc[r] + bias);
  }
}

// ---------------- kernel 3: persistent time-scan (4 batches x 10 slots per WG) ----------------
__global__ __launch_bounds__(NTHR) void scan_kernel(
    const float* __restrict__ inputs, const float* __restrict__ yv,
    const int* __restrict__ klg, const float* __restrict__ Wts,
    const float* __restrict__ bg, const float* __restrict__ battn,
    const unsigned short* __restrict__ qp, unsigned short* __restrict__ pk,
    unsigned short* __restrict__ pv, const unsigned int* __restrict__ wpack,
    float* __restrict__ out) {
  extern __shared__ unsigned char smem[];
  float* hbuf = (float*)(smem + OFF_HBUF);
  float* cbuf = (float*)(smem + OFF_CBUF);
  float* ph   = (float*)(smem + OFF_PH);
  float* px   = (float*)(smem + OFF_PX);
  float* ctx  = (float*)(smem + OFF_CTX);
  float* atb  = (float*)(smem + OFF_AT);
  float* sc   = (float*)(smem + OFF_SC);
  unsigned short* abuf = (unsigned short*)(smem + OFF_ABUF);
  unsigned short* xtb  = (unsigned short*)(smem + OFF_XTB);
  int*   nv   = (int*)(smem + OFF_NV);
  float* lsum = (float*)(smem + OFF_LSUM);
  float* sTt  = (float*)(smem + OFF_STT);
  float* sDt  = (float*)(smem + OFF_SDT);
  float* sy   = (float*)(smem + OFF_SY);
  int*   skl  = (int*)(smem + OFF_SKL);

  const int tid = threadIdx.x, lane = tid & 31, wave = tid >> 5;
  const int b0 = blockIdx.x * NB;
  const float scale = 0.08838834764831845f;  // 1/sqrt(128)
  float lacc = 0.0f;

  // ---- init: mask histogram nv(t), state zero, LDS pad rows zero ----
  if (tid < NB) skl[tid] = klg[b0 + tid];
  if (tid < T_) {
    int c = 0;
    for (int b = 0; b < B_; ++b) c += (tid < klg[b]) ? 1 : 0;
    nv[tid] = c;
  }
  for (int i = tid; i < NROW * H_; i += NTHR) { hbuf[i] = 0.0f; cbuf[i] = 0.0f; }
  for (int i = tid; i < 8 * ASTRIDE; i += NTHR)  abuf[NROW * ASTRIDE + i] = 0;  // rows 40..47
  for (int i = tid; i < 12 * ASTRIDE; i += NTHR) xtb[NB * ASTRIDE + i] = 0;     // rows 4..15
  __syncthreads();

  for (int t = 0; t < T_; ++t) {
    // ---- phase 1: scalars, xt stage (bf16), h stage (bf16) ----
    if (tid < NB) {
      const float* row = inputs + ((size_t)(b0 + tid) * T_ + t) * (2 + IPT_);
      sTt[tid] = row[0];
      sDt[tid] = row[1];
      sy[tid]  = yv[(size_t)(b0 + tid) * T_ + t];
    }
    if (tid >= NB && tid < 2 * NB) lsum[tid - NB] = 0.0f;
    {
      int bi = tid >> 7, k = tid & 127;  // 512 threads == NB*128
      const float* row = inputs + ((size_t)(b0 + bi) * T_ + t) * (2 + IPT_) + 2;
      xtb[bi * ASTRIDE + k] = f2bf(row[k]);
    }
    for (int i = tid; i < NROW * H_; i += NTHR) {
      int r = i >> 7, hh = i & 127;
      abuf[r * ASTRIDE + hh] = f2bf(hbuf[i]);
    }
    __syncthreads();

    // ---- phase 2: ph = h@Wh[g] (96 tiles) and px = xt@Wx[g] (64 tiles) via WMMA ----
    for (int idx = wave; idx < 160; idx += 16) {
      if (idx < 96) {
        int g = idx / 24, rem = idx % 24, mt = rem >> 3, nt = rem & 7;
        v8f acc = wmma_gemm_128(abuf, ASTRIDE, mt * 16, wpack + (size_t)(5 + g) * 8192, nt, lane);
        store_tile(ph + (size_t)g * 5120, H_, mt * 16, NROW, nt * 16, acc, lane);
      } else {
        int j = idx - 96, g = j >> 3, nt = j & 7;
        v8f acc = wmma_gemm_128(xtb, ASTRIDE, 0, wpack + (size_t)(9 + g) * 8192, nt, lane);
        store_tile(px + (size_t)g * 512, H_, 0, NB, nt * 16, acc, lane);
      }
    }
    __syncthreads();

    // ---- phase 3: causal attention over pk/pv history (VALU, bf16 from L2) ----
    if (t > 0) {
      for (int i = tid; i < NROW * t; i += NTHR) {
        int r = i / t, tp = i - r * t;
        int bi = r / S_, s = r - bi * S_, b = b0 + bi;
        const unsigned int* qrow = (const unsigned int*)(qp + (((size_t)b * T_ + t) * S_ + s) * H_);
        const unsigned int* krow = (const unsigned int*)(pk + (((size_t)b * T_ + tp) * S_ + s) * H_);
        float acc = 0.0f;
#pragma unroll 8
        for (int k = 0; k < H_ / 2; ++k) {
          unsigned int qu = qrow[k], ku = krow[k];
          acc += bflo(qu) * bflo(ku) + bfhi(qu) * bfhi(ku);
        }
        sc[r * 64 + tp] = acc * scale;
      }
      __syncthreads();
      if (tid < NROW) {  // per-row softmax over t' in [0, t)
        float mx = -1e30f;
        for (int tp = 0; tp < t; ++tp) mx = fmaxf(mx, sc[tid * 64 + tp]);
        float sum = 0.0f;
        for (int tp = 0; tp < t; ++tp) {
          float e = __expf(sc[tid * 64 + tp] - mx);
          sc[tid * 64 + tp] = e;
          sum += e;
        }
        float inv = 1.0f / sum;
        for (int tp = 0; tp < t; ++tp) sc[tid * 64 + tp] *= inv;
      }
      __syncthreads();
      for (int i = tid; i < NROW * H_; i += NTHR) {  // ctx = prob @ pv
        int r = i >> 7, hh = i & 127;
        int bi = r / S_, s = r - bi * S_, b = b0 + bi;
        const unsigned short* pvb = pv + ((size_t)b * T_ * S_ + s) * H_ + hh;
        float acc = 0.0f;
        for (int tp = 0; tp < t; ++tp) acc += sc[r * 64 + tp] * bf2f(pvb[(size_t)tp * S_ * H_]);
        ctx[i] = acc;
      }
      __syncthreads();
      for (int i = tid; i < NROW * H_; i += NTHR) {  // stage ctx -> bf16
        int r = i >> 7, hh = i & 127;
        abuf[r * ASTRIDE + hh] = f2bf(ctx[i]);
      }
      __syncthreads();
      for (int idx = wave; idx < 24; idx += 16) {  // at = ctx @ Wattn[3]
        int mt = idx >> 3, nt = idx & 7;
        v8f acc = wmma_gemm_128(abuf, ASTRIDE, mt * 16, wpack + (size_t)4 * 8192, nt, lane);
        store_tile(atb, H_, mt * 16, NROW, nt * 16, acc, lane);
      }
    }
    __syncthreads();

    // ---- phase 4: gates, cell/hidden update, hseq out, BCE partials ----
    for (int i = tid; i < NROW * H_; i += NTHR) {
      int r = i >> 7, hh = i & 127;
      int bi = r / S_, s = r - bi * S_, b = b0 + bi;
      float tt = sTt[bi], dd = sDt[bi];
      const float* pxb = px + (size_t)bi * H_ + hh;
      float p0 = pxb[0], p1 = pxb[512], p2 = pxb[2 * 512], p3 = pxb[3 * 512];
      float p4 = pxb[4 * 512], p5 = pxb[5 * 512], p6 = pxb[6 * 512], p7 = pxb[7 * 512];
      float itg = sigmf(p0 + ph[i] + bg[hh]);
      float ftg = sigmf(p1 + ph[5120 + i] + bg[H_ + hh]);
      float ctl = tanhf(p2 + ph[2 * 5120 + i] + bg[2 * H_ + hh]);
      float otg = sigmf(p3 + ph[3 * 5120 + i] + tt * Wts[4 * H_ + hh] + dd * Wts[5 * H_ + hh] + bg[3 * H_ + hh]);
      float T1g = sigmf(p4 + sigmf(tt * Wts[hh]) + bg[4 * H_ + hh]);
      float T2g = sigmf(p5 + sigmf(tt * Wts[H_ + hh]) + bg[5 * H_ + hh]);
      float D1g = sigmf(p6 + sigmf(dd * Wts[2 * H_ + hh]) + bg[6 * H_ + hh]);
      float D2g = sigmf(p7 + sigmf(dd * Wts[3 * H_ + hh]) + bg[7 * H_ + hh]);
      float atv = (t == 0) ? 1.0f : (atb[i] + battn[3 * H_ + hh]);
      float itp = itg * atv, ftp = ftg * atv;
      float cold = cbuf[i], hold = hbuf[i];
      float base = ftp * cold;
      float chat = base + itp * (T1g * D1g) * ctl;
      float cnew = base + itp * (T2g * D2g) * ctl;
      float hnew = otg * tanhf(chat);
      bool msk = (t < skl[bi]);
      hbuf[i] = msk ? hnew : hold;
      cbuf[i] = msk ? cnew : cold;
      float hw = msk ? hnew : 0.0f;
      out[(((size_t)b * T_ + t) * S_ + s) * H_ + hh] = hw;    // hidden_seq_attn
      abuf[r * ASTRIDE + hh] = f2bf(hw);
      float pcl = fminf(fmaxf(hnew, 1e-7f), 1.0f - 1e-7f);
      float yb = sy[bi];
      float bce = -(yb * __logf(pcl) + (1.0f - yb) * log1pf(-pcl));
      atomicAdd(&lsum[bi], bce);                              // ds_add_f32
    }
    __syncthreads();

    // ---- phase 5: pk/pv projections of h_w (WMMA, reusing ph LDS) + loss step ----
    float* pko = ph;
    float* pvo = ph + 5120;
    for (int idx = wave; idx < 48; idx += 16) {
      int which = idx / 24, rem = idx % 24, mt = rem >> 3, nt = rem & 7;
      v8f acc = wmma_gemm_128(abuf, ASTRIDE, mt * 16, wpack + (size_t)(2 + which) * 8192, nt, lane);
      store_tile(which ? pvo : pko, H_, mt * 16, NROW, nt * 16, acc, lane);
    }
    if (tid == 0) {
      int n = nv[t];
      if (n > 0) {
        for (int bi = 0; bi < NB; ++bi)
          if (t < skl[bi]) lacc += lsum[bi] * (1.0f / (S_ * H_)) / (float)n;
      }
    }
    __syncthreads();

    // ---- phase 6: bias + bf16 store of pk/pv rows for step t ----
    for (int i = tid; i < NROW * H_; i += NTHR) {
      int r = i >> 7, hh = i & 127;
      int bi = r / S_, s = r - bi * S_, b = b0 + bi;
      size_t off = (((size_t)b * T_ + t) * S_ + s) * H_ + hh;
      pk[off] = f2bf(pko[i] + battn[H_ + hh]);
      pv[off] = f2bf(pvo[i] + battn[2 * H_ + hh]);
    }
    __syncthreads();
  }

  // ---- epilogue: final h, c, loss ----
  for (int i = tid; i < NROW * H_; i += NTHR) {
    int r = i >> 7, hh = i & 127;
    int bi = r / S_, s = r - bi * S_, b = b0 + bi;
    out[OUT_H + ((size_t)b * S_ + s) * H_ + hh] = hbuf[i];
    out[OUT_C + ((size_t)b * S_ + s) * H_ + hh] = cbuf[i];
  }
  if (tid == 0) atomicAdd(out + OUT_LOSS, lacc);
}

// ---------------- host launcher ----------------
extern "C" void kernel_launch(void* const* d_in, const int* in_sizes, int n_in,
                              void* d_out, int out_size, void* d_ws, size_t ws_size,
                              hipStream_t stream) {
  (void)in_sizes; (void)n_in; (void)out_size; (void)ws_size;
  const float* inputs = (const float*)d_in[0];
  const float* x_q    = (const float*)d_in[1];
  const float* yv     = (const float*)d_in[2];
  const int*   kl     = (const int*)d_in[3];
  const float* Wx     = (const float*)d_in[4];
  const float* Wh     = (const float*)d_in[5];
  const float* Wts    = (const float*)d_in[6];
  const float* bg     = (const float*)d_in[7];
  const float* Wa     = (const float*)d_in[8];
  const float* ba     = (const float*)d_in[9];
  const float* Wattn  = (const float*)d_in[10];
  const float* battn  = (const float*)d_in[11];
  float* out = (float*)d_out;

  unsigned char* ws = (unsigned char*)d_ws;
  unsigned int*   wpack = (unsigned int*)(ws);
  unsigned short* qp    = (unsigned short*)(ws + QP_OFF);
  unsigned short* pk    = (unsigned short*)(ws + PK_OFF);
  unsigned short* pv    = (unsigned short*)(ws + PV_OFF);

  // 1) pack weights to WMMA B-fragment layout (bf16), zero loss slot
  pack_kernel<<<(WPACK_UINTS + 255u) / 256u, 256, 0, stream>>>(Wa, Wattn, Wh, Wx, wpack, out);

  // 2) qp precompute: 163840 rows, 16-row tile per WG (8 waves = 8 N-tiles)
  qp_kernel<<<(B_ * T_ * S_) / 16, 256, 0, stream>>>(x_q, ba, battn, wpack, qp);

  // 3) persistent scan: 64 WGs x 512 threads, ~208KB LDS each (one WG per WGP)
  hipFuncSetAttribute((const void*)scan_kernel,
                      hipFuncAttributeMaxDynamicSharedMemorySize, (int)SMEM_SCAN);
  scan_kernel<<<B_ / NB, NTHR, SMEM_SCAN, stream>>>(inputs, yv, kl, Wts, bg, battn,
                                                    qp, pk, pv, wpack, out);
}
